// MixedRepeatHeads_64476049047750
// MI455X (gfx1250) — compile-verified, matrix operand order checked by reading, not run on hardware
//
#include <hip/hip_runtime.h>
#include <hip/hip_bf16.h>

// ---- problem sizes (match reference) ----
#define B_SZ   8192
#define DIM    4096
#define HID    256
#define NHEAD  16
#define SEQQ   2048
#define NCOLS  4096          // H*HID
// ---- tiling ----
#define BM 128
#define BN 128
#define BK 32
#define NT (DIM / BK)        // 128 K-iterations
#define PITCH 40             // LDS row pitch in bf16 elems (32 data + 8 pad -> 80B, conflict-free)
#define TILE_ELEMS (BM * PITCH)

typedef __attribute__((ext_vector_type(16))) __bf16 v16bf;
typedef __attribute__((ext_vector_type(8)))  __bf16 v8bf;
typedef __attribute__((ext_vector_type(8)))  float  v8f;

union FragU { v16bf v; v8bf h[2]; };

// A/B fragment load per the CDNA5 16-bit operand layout:
// lane L -> row (L%16); K chunk (L>>4)*8; VGPR0-3 = K[chunk..chunk+8),
// VGPR4-7 = K[16+chunk..16+chunk+8)  => two 16B LDS loads at +0 / +32 bytes.
__device__ __forceinline__ v16bf load_frag(const __bf16* tile, int row, int lane) {
    const int chunk = (lane >> 4) << 3;   // 0 or 8 elems
    const __bf16* p = tile + row * PITCH + chunk;
    FragU u;
    u.h[0] = *(const v8bf*)(p);
    u.h[1] = *(const v8bf*)(p + 16);
    return u.v;
}

__device__ __forceinline__ v8bf cvt8(const float* s) {
    v8bf r;
#pragma unroll
    for (int i = 0; i < 8; ++i) r[i] = (__bf16)s[i];
    return r;
}

// Async global->LDS 16B copy (CDNA5): LDS dest address in VDST VGPR,
// 64-bit global address in VGPR pair, tracked on ASYNCcnt.
__device__ __forceinline__ void async_b128(const __bf16* g, __bf16* l) {
    unsigned loff = (unsigned)(uintptr_t)l;   // low 32 bits of generic ptr = LDS offset
    asm volatile("global_load_async_to_lds_b128 %0, %1, off"
                 :: "v"(loff), "v"(g) : "memory");
}
__device__ __forceinline__ void wait_async0() {
    asm volatile("s_wait_asynccnt 0x0" ::: "memory");
}

// ---------------- GEMM1: proj = x @ W_proj[h], fused mix epilogue -> hidden(bf16) ----------------
__global__ __launch_bounds__(256) void k_proj_mix(
    const float* __restrict__ x, const float* __restrict__ Wp,
    const float* __restrict__ bp, const float* __restrict__ wmix,
    const float* __restrict__ bmix, const float* __restrict__ dvals,
    const float* __restrict__ caches, const int* __restrict__ idxp,
    __bf16* __restrict__ hidden)
{
    __shared__ __bf16 ldsA[2 * TILE_ELEMS];
    __shared__ __bf16 ldsB[2 * TILE_ELEMS];

    const int n0 = blockIdx.x * BN;
    const int m0 = blockIdx.y * BM;
    const int h  = n0 >> 8;                  // head uniform per block (BN=128 <= HID=256)

    const int tid  = threadIdx.x;
    const int lane = tid & 31;
    const int ln   = lane & 15;
    const int wv   = tid >> 5;               // 8 waves
    const int wm   = wv >> 2;                // 0..1  (64-row slab)
    const int wn   = wv & 3;                 // 0..3  (32-col slab)

    // loader coords: each thread owns one LDS row-half (16 K values)
    const int r  = tid & 127;
    const int kc = (tid >> 7) * 16;

    // per-head scalars (reference: w = w_mix[:,index]; decay = clip(dv,.9,1)^(1/4))
    const int index   = idxp[0];
    const float wcoef = wmix[h * SEQQ + index];
    const float bcoef = bmix[h * SEQQ + index];
    float d = dvals[h];
    d = fminf(fmaxf(d, 0.9f), 1.0f);
    d = __powf(d, 0.25f);
    const float ccoef = (h < (NHEAD / 2)) ? wcoef * d : d;

    v8f acc[4][2];
#pragma unroll
    for (int i = 0; i < 4; ++i)
#pragma unroll
        for (int j = 0; j < 2; ++j)
#pragma unroll
            for (int v = 0; v < 8; ++v) acc[i][j][v] = 0.0f;

    const int    kcol  = (n0 + r) & (HID - 1);
    const float* bsrc0 = Wp + (size_t)h * DIM * HID + kcol;
    const float* asrc0 = x + (size_t)(m0 + r) * DIM + kc;

    float pa[16], pb[16];

    // ---- prologue: fetch tile 0 and stage into buffer 0 ----
    {
        const float4* p4 = (const float4*)(asrc0);
#pragma unroll
        for (int q = 0; q < 4; ++q) {
            float4 t = p4[q];
            pa[4*q+0] = t.x; pa[4*q+1] = t.y; pa[4*q+2] = t.z; pa[4*q+3] = t.w;
        }
        const float* src = bsrc0 + (size_t)kc * HID;
#pragma unroll
        for (int q = 0; q < 16; ++q) pb[q] = src[(size_t)q * HID];
        *(v8bf*)(ldsA + r * PITCH + kc)     = cvt8(pa);
        *(v8bf*)(ldsA + r * PITCH + kc + 8) = cvt8(pa + 8);
        *(v8bf*)(ldsB + r * PITCH + kc)     = cvt8(pb);
        *(v8bf*)(ldsB + r * PITCH + kc + 8) = cvt8(pb + 8);
    }
    __syncthreads();

    for (int kt = 0; kt < NT; ++kt) {
        const int cur = kt & 1;
        const __bf16* As = ldsA + cur * TILE_ELEMS;
        const __bf16* Bs = ldsB + cur * TILE_ELEMS;

        // issue next tile's global loads early (latency hidden under WMMA phase)
        if (kt + 1 < NT) {
            const float4* p4 = (const float4*)(asrc0 + (kt + 1) * BK);
#pragma unroll
            for (int q = 0; q < 4; ++q) {
                float4 t = p4[q];
                pa[4*q+0] = t.x; pa[4*q+1] = t.y; pa[4*q+2] = t.z; pa[4*q+3] = t.w;
            }
            const float* src = bsrc0 + (size_t)((kt + 1) * BK + kc) * HID;
#pragma unroll
            for (int q = 0; q < 16; ++q) pb[q] = src[(size_t)q * HID];
        }

        // ---- WMMA phase on current buffer ----
        v16bf af[4], bf[2];
#pragma unroll
        for (int i = 0; i < 4; ++i) af[i] = load_frag(As, wm * 64 + i * 16 + ln, lane);
#pragma unroll
        for (int j = 0; j < 2; ++j) bf[j] = load_frag(Bs, wn * 32 + j * 16 + ln, lane);
#pragma unroll
        for (int i = 0; i < 4; ++i)
#pragma unroll
            for (int j = 0; j < 2; ++j)
                acc[i][j] = __builtin_amdgcn_wmma_f32_16x16x32_bf16(
                    false, af[i], false, bf[j], (short)0, acc[i][j], false, false);

        // stage next tile into the other buffer
        if (kt + 1 < NT) {
            __bf16* An = ldsA + (cur ^ 1) * TILE_ELEMS;
            __bf16* Bn = ldsB + (cur ^ 1) * TILE_ELEMS;
            *(v8bf*)(An + r * PITCH + kc)     = cvt8(pa);
            *(v8bf*)(An + r * PITCH + kc + 8) = cvt8(pa + 8);
            *(v8bf*)(Bn + r * PITCH + kc)     = cvt8(pb);
            *(v8bf*)(Bn + r * PITCH + kc + 8) = cvt8(pb + 8);
        }
        __syncthreads();
    }

    // ---- fused epilogue: hidden = w*(proj + b_proj) + coef*caches + b, as bf16 ----
    const int lh = lane >> 4;                // C layout: VGPR v -> M = v + lh*8
    const float* cbase = caches + (size_t)h * B_SZ * HID;
#pragma unroll
    for (int j = 0; j < 2; ++j) {
        const int gn  = n0 + wn * 32 + j * 16 + ln;
        const int k   = gn & (HID - 1);
        const float bpv = bp[h * HID + k];
#pragma unroll
        for (int i = 0; i < 4; ++i) {
#pragma unroll
            for (int v = 0; v < 8; ++v) {
                const int gm = m0 + wm * 64 + i * 16 + v + lh * 8;
                const float c     = acc[i][j][v];
                // caches is streamed exactly once -> non-temporal, keep L2 for x/W/hidden
                const float cache = __builtin_nontemporal_load(cbase + (size_t)gm * HID + k);
                const float val   = wcoef * (c + bpv) + ccoef * cache + bcoef;
                hidden[(size_t)gm * NCOLS + gn] = (__bf16)val;
            }
        }
    }
}

// ---------------- GEMM2: out = hidden(bf16) @ W_out + b_out, fp32 out ----------------
__global__ __launch_bounds__(256) void k_out(
    const __bf16* __restrict__ hidden, const float* __restrict__ Wout,
    const float* __restrict__ bout, float* __restrict__ out)
{
    __shared__ __bf16 ldsA[2 * TILE_ELEMS];
    __shared__ __bf16 ldsB[2 * TILE_ELEMS];

    const int n0 = blockIdx.x * BN;
    const int m0 = blockIdx.y * BM;

    const int tid  = threadIdx.x;
    const int lane = tid & 31;
    const int ln   = lane & 15;
    const int wv   = tid >> 5;
    const int wm   = wv >> 2;
    const int wn   = wv & 3;

    const int r  = tid & 127;
    const int kc = (tid >> 7) * 16;

    v8f acc[4][2];
#pragma unroll
    for (int i = 0; i < 4; ++i)
#pragma unroll
        for (int j = 0; j < 2; ++j)
#pragma unroll
            for (int v = 0; v < 8; ++v) acc[i][j][v] = 0.0f;

    const __bf16* asrc0 = hidden + (size_t)(m0 + r) * NCOLS + kc;
    const float*  bsrc0 = Wout + (size_t)kc * DIM + n0 + r;

    float pb[16];

    // ---- prologue: async-DMA A tile 0 into buffer 0, stage B tile 0 ----
    {
        __bf16* Al = ldsA + r * PITCH + kc;
        async_b128(asrc0,     Al);
        async_b128(asrc0 + 8, Al + 8);
        const float* src = bsrc0;
#pragma unroll
        for (int q = 0; q < 16; ++q) pb[q] = src[(size_t)q * DIM];
        *(v8bf*)(ldsB + r * PITCH + kc)     = cvt8(pb);
        *(v8bf*)(ldsB + r * PITCH + kc + 8) = cvt8(pb + 8);
    }
    wait_async0();
    __syncthreads();

    for (int kt = 0; kt < NT; ++kt) {
        const int cur = kt & 1;
        const __bf16* As = ldsA + cur * TILE_ELEMS;
        const __bf16* Bs = ldsB + cur * TILE_ELEMS;

        // issue next A tile as async DMA straight into the other LDS buffer
        // (no VGPR staging; overlaps the WMMA phase, drained on ASYNCcnt)
        if (kt + 1 < NT) {
            __bf16* An = ldsA + (cur ^ 1) * TILE_ELEMS + r * PITCH + kc;
            const __bf16* g = asrc0 + (kt + 1) * BK;
            async_b128(g,     An);
            async_b128(g + 8, An + 8);
            const float* src = bsrc0 + (size_t)(kt + 1) * BK * DIM;
#pragma unroll
            for (int q = 0; q < 16; ++q) pb[q] = src[(size_t)q * DIM];
        }

        // ---- WMMA phase on current buffer ----
        v16bf af[4], bf[2];
#pragma unroll
        for (int i = 0; i < 4; ++i) af[i] = load_frag(As, wm * 64 + i * 16 + ln, lane);
#pragma unroll
        for (int j = 0; j < 2; ++j) bf[j] = load_frag(Bs, wn * 32 + j * 16 + ln, lane);
#pragma unroll
        for (int i = 0; i < 4; ++i)
#pragma unroll
            for (int j = 0; j < 2; ++j)
                acc[i][j] = __builtin_amdgcn_wmma_f32_16x16x32_bf16(
                    false, af[i], false, bf[j], (short)0, acc[i][j], false, false);

        if (kt + 1 < NT) {
            __bf16* Bn = ldsB + (cur ^ 1) * TILE_ELEMS;
            *(v8bf*)(Bn + r * PITCH + kc)     = cvt8(pb);
            *(v8bf*)(Bn + r * PITCH + kc + 8) = cvt8(pb + 8);
        }
        wait_async0();       // A-tile DMA for next iteration complete
        __syncthreads();
    }

    const int lh = lane >> 4;
#pragma unroll
    for (int j = 0; j < 2; ++j) {
        const int gn = n0 + wn * 32 + j * 16 + ln;
        const float bo = bout[gn];
#pragma unroll
        for (int i = 0; i < 4; ++i) {
#pragma unroll
            for (int v = 0; v < 8; ++v) {
                const int gm = m0 + wm * 64 + i * 16 + v + lh * 8;
                out[(size_t)gm * DIM + gn] = acc[i][j][v] + bo;
            }
        }
    }
}

extern "C" void kernel_launch(void* const* d_in, const int* in_sizes, int n_in,
                              void* d_out, int out_size, void* d_ws, size_t ws_size,
                              hipStream_t stream) {
    const float* x      = (const float*)d_in[0];
    const int*   idx    = (const int*)  d_in[1];
    const float* Wp     = (const float*)d_in[2];
    const float* bp     = (const float*)d_in[3];
    const float* Wout   = (const float*)d_in[4];
    const float* bout   = (const float*)d_in[5];
    const float* wmix   = (const float*)d_in[6];
    const float* bmix   = (const float*)d_in[7];
    const float* dvals  = (const float*)d_in[8];
    const float* caches = (const float*)d_in[9];
    float*   out    = (float*)d_out;
    __bf16*  hidden = (__bf16*)d_ws;   // 8192*4096*2 = 64 MB scratch

    dim3 block(256);
    dim3 grid(NCOLS / BN, B_SZ / BM);  // 32 x 64

    k_proj_mix<<<grid, block, 0, stream>>>(x, Wp, bp, wmix, bmix, dvals, caches, idx, hidden);
    k_out<<<grid, block, 0, stream>>>(hidden, Wout, bout, out);
}
